// ActorNetwork_67662914781471
// MI455X (gfx1250) — matrix operand order repaired and checked by reading, not compiled, not run on hardware
//
#include <hip/hip_runtime.h>
#include <math.h>

// ---- problem constants (match reference) ----
#define N_NODES 5000
#define F_DIM   128
#define H_HEADS 4
#define C_DIM   32
#define HC_DIM  128
#define GH_DIM  256
#define E_EDGES 320000
#define ETOT    (E_EDGES + N_NODES)
#define NEG_SLOPE 0.2f

typedef float  v2f  __attribute__((ext_vector_type(2)));
typedef float  v8f  __attribute__((ext_vector_type(8)));
typedef __bf16 bf16_t;
typedef __bf16 v16bf __attribute__((ext_vector_type(16)));

// ---------------- float atomic max (CAS loop) ----------------
__device__ inline void atomic_max_f32(float* addr, float val) {
    unsigned int* ia = (unsigned int*)addr;
    unsigned int cur = __float_as_uint(*addr);
    while (__uint_as_float(cur) < val) {
        unsigned int prev = atomicCAS(ia, cur, __float_as_uint(val));
        if (prev == cur) break;
        cur = prev;
    }
}

// ---------------- init: emax=-inf, denom=0, agg=0 ----------------
__global__ void k_init(float* emax, float* denom, float* agg) {
    int t = blockIdx.x * blockDim.x + threadIdx.x;
    if (t < N_NODES * H_HEADS) { emax[t] = -INFINITY; denom[t] = 0.0f; }
    if (t < N_NODES * HC_DIM)  { agg[t]  = 0.0f; }
}

// ---------------- generic fp32 WMMA GEMM (exact, for small GEMMs) ----------------
// C[M,Nn] = A[M,K] @ (TRANS_B ? B[Nn,K]^T : B[K,Nn]) + bias[Nn]
// 8 waves/block, block tile 64x64, wave tile 16x32. Indices clamped (not
// predicated): OOB garbage lands only in store-guarded elements.
template <bool TRANS_B>
__global__ __launch_bounds__(256) void k_gemm_wmma_f32(
        const float* __restrict__ A, const float* __restrict__ B,
        const float* __restrict__ bias, float* __restrict__ C,
        int M, int Nn, int K) {
    const int lane = threadIdx.x & 31;
    const int wave = threadIdx.x >> 5;
    const int m0 = blockIdx.x * 64 + (wave >> 1) * 16;
    const int n0 = blockIdx.y * 64 + (wave & 1) * 32;
    const int ml = lane & 15;
    const int kb = (lane >> 4) << 1;   // 0 or 2

    const int am  = min(m0 + ml,      M - 1);
    const int bn0 = min(n0 + ml,      Nn - 1);
    const int bn1 = min(n0 + 16 + ml, Nn - 1);

    const float* __restrict__ Arow = A + (size_t)am * K + kb;
    const float* __restrict__ Brow0;
    const float* __restrict__ Brow1;
    if (TRANS_B) {
        Brow0 = B + (size_t)bn0 * K + kb;
        Brow1 = B + (size_t)bn1 * K + kb;
    } else {
        Brow0 = B + (size_t)kb * Nn + bn0;
        Brow1 = B + (size_t)kb * Nn + bn1;
    }

    v8f acc0 = {}, acc1 = {};

#pragma unroll 4
    for (int k = 0; k < K; k += 4) {
        v2f a = *(const v2f*)(Arow + k);
        v2f b0, b1;
        if (TRANS_B) {
            b0 = *(const v2f*)(Brow0 + k);
            b1 = *(const v2f*)(Brow1 + k);
        } else {
            b0.x = Brow0[(size_t)k * Nn];
            b0.y = Brow0[(size_t)k * Nn + Nn];
            b1.x = Brow1[(size_t)k * Nn];
            b1.y = Brow1[(size_t)k * Nn + Nn];
        }
        acc0 = __builtin_amdgcn_wmma_f32_16x16x4_f32(false, a, false, b0,
                                                     (short)0, acc0, false, false);
        acc1 = __builtin_amdgcn_wmma_f32_16x16x4_f32(false, a, false, b1,
                                                     (short)0, acc1, false, false);
    }

    const int col0 = n0 + ml;
    const int col1 = col0 + 16;
    const float bia0 = (col0 < Nn) ? bias[col0] : 0.0f;
    const float bia1 = (col1 < Nn) ? bias[col1] : 0.0f;

#pragma unroll
    for (int r = 0; r < 8; ++r) {
        int row = m0 + r + ((lane >> 4) << 3);
        if (row < M) {
            if (col0 < Nn) C[(size_t)row * Nn + col0] = acc0[r] + bia0;
            if (col1 < Nn) C[(size_t)row * Nn + col1] = acc1[r] + bia1;
        }
    }
}

// ---------------- bf16 split: v = hi + lo (each bf16, RNE) ----------------
__global__ void k_split_bf16(const float* __restrict__ src,
                             bf16_t* __restrict__ hi, bf16_t* __restrict__ lo,
                             int n) {
    int t = blockIdx.x * blockDim.x + threadIdx.x;
    if (t >= n) return;
    float v  = src[t];
    bf16_t h = (bf16_t)v;
    hi[t] = h;
    lo[t] = (bf16_t)(v - (float)h);
}

// ---------------- bf16x3 NT GEMM (policy head) ----------------
// C[M,Nn] = (Ahi+Alo)[M,K] @ (Bhi+Blo)[Nn,K]^T + bias  ~  Ahi*Bhi + Ahi*Blo + Alo*Bhi
// fp32 accumulate via V_WMMA_F32_16X16X32_BF16. Fragment layouts (ISA 7.12.2):
//   A 16x32 (M=lane&15, half=lane>>4): elems 0-7  <- K = half*8..half*8+7
//                                      elems 8-15 <- K = 16+half*8..16+half*8+7
//   B 32x16 (N=lane&15, half=lane>>4): same K mapping as A
// All per-lane chunks are 16B-aligned contiguous runs -> b128 loads.
__device__ inline v16bf ld_a_frag(const bf16_t* row, int k, int half) {
    union { uint4 u[2]; v16bf v; } t;
    t.u[0] = *(const uint4*)(row + k + half * 8);
    t.u[1] = *(const uint4*)(row + k + 16 + half * 8);
    return t.v;
}
__device__ inline v16bf ld_b_frag(const bf16_t* row, int k, int half) {
    union { uint4 u[2]; v16bf v; } t;
    t.u[0] = *(const uint4*)(row + k + half * 8);
    t.u[1] = *(const uint4*)(row + k + 16 + half * 8);
    return t.v;
}

__global__ __launch_bounds__(256) void k_gemm_bf16x3_nt(
        const bf16_t* __restrict__ Ahi, const bf16_t* __restrict__ Alo,
        const bf16_t* __restrict__ Bhi, const bf16_t* __restrict__ Blo,
        const float* __restrict__ bias, float* __restrict__ C,
        int M, int Nn, int K) {
    const int lane = threadIdx.x & 31;
    const int wave = threadIdx.x >> 5;
    const int m0 = blockIdx.x * 64 + (wave >> 1) * 16;
    const int n0 = blockIdx.y * 64 + (wave & 1) * 32;
    const int ml   = lane & 15;
    const int half = lane >> 4;

    const int am  = min(m0 + ml,      M - 1);
    const int bn0 = min(n0 + ml,      Nn - 1);
    const int bn1 = min(n0 + 16 + ml, Nn - 1);

    const bf16_t* Ah  = Ahi + (size_t)am  * K;
    const bf16_t* Al  = Alo + (size_t)am  * K;
    const bf16_t* B0h = Bhi + (size_t)bn0 * K;
    const bf16_t* B0l = Blo + (size_t)bn0 * K;
    const bf16_t* B1h = Bhi + (size_t)bn1 * K;
    const bf16_t* B1l = Blo + (size_t)bn1 * K;

    v8f acc0 = {}, acc1 = {};

#pragma unroll 2
    for (int k = 0; k < K; k += 32) {
        v16bf ah  = ld_a_frag(Ah,  k, half);
        v16bf al  = ld_a_frag(Al,  k, half);
        v16bf b0h = ld_b_frag(B0h, k, half);
        v16bf b0l = ld_b_frag(B0l, k, half);
        v16bf b1h = ld_b_frag(B1h, k, half);
        v16bf b1l = ld_b_frag(B1l, k, half);

        acc0 = __builtin_amdgcn_wmma_f32_16x16x32_bf16(false, ah, false, b0h,
                                                       (short)0, acc0, false, false);
        acc1 = __builtin_amdgcn_wmma_f32_16x16x32_bf16(false, ah, false, b1h,
                                                       (short)0, acc1, false, false);
        acc0 = __builtin_amdgcn_wmma_f32_16x16x32_bf16(false, ah, false, b0l,
                                                       (short)0, acc0, false, false);
        acc1 = __builtin_amdgcn_wmma_f32_16x16x32_bf16(false, ah, false, b1l,
                                                       (short)0, acc1, false, false);
        acc0 = __builtin_amdgcn_wmma_f32_16x16x32_bf16(false, al, false, b0h,
                                                       (short)0, acc0, false, false);
        acc1 = __builtin_amdgcn_wmma_f32_16x16x32_bf16(false, al, false, b1h,
                                                       (short)0, acc1, false, false);
    }

    const int col0 = n0 + ml;
    const int col1 = col0 + 16;
    const float bia0 = (col0 < Nn) ? bias[col0] : 0.0f;
    const float bia1 = (col1 < Nn) ? bias[col1] : 0.0f;

#pragma unroll
    for (int r = 0; r < 8; ++r) {
        int row = m0 + r + ((lane >> 4) << 3);
        if (row < M) {
            if (col0 < Nn)
                __builtin_nontemporal_store(acc0[r] + bia0, &C[(size_t)row * Nn + col0]);
            if (col1 < Nn)
                __builtin_nontemporal_store(acc1[r] + bia1, &C[(size_t)row * Nn + col1]);
        }
    }
}

// ---------------- edge kernels (1 thread = 1 (edge, head)) ----------------
__device__ inline void edge_endpoints(int e, const int* __restrict__ ei,
                                      int& src, int& dst) {
    if (e < E_EDGES) { src = ei[e]; dst = ei[E_EDGES + e]; }
    else             { src = e - E_EDGES; dst = e - E_EDGES; }  // self loops
}

__global__ void k_edge_score(const int* __restrict__ ei,
                             const float* __restrict__ xl,
                             const float* __restrict__ xr,
                             const float* __restrict__ att,
                             float* __restrict__ escore,
                             float* __restrict__ emax) {
    int t = blockIdx.x * blockDim.x + threadIdx.x;
    if (t >= ETOT * H_HEADS) return;
    int e = t >> 2, h = t & 3;
    int src, dst; edge_endpoints(e, ei, src, dst);
    const float4* xls = (const float4*)(xl + (size_t)src * HC_DIM + h * C_DIM);
    const float4* xrs = (const float4*)(xr + (size_t)dst * HC_DIM + h * C_DIM);
    const float4* at  = (const float4*)(att + h * C_DIM);
    float s = 0.0f;
#pragma unroll
    for (int c4 = 0; c4 < C_DIM / 4; ++c4) {
        float4 l = xls[c4], r = xrs[c4], a = at[c4];
        float m;
        m = l.x + r.x; m = (m > 0.0f) ? m : NEG_SLOPE * m; s += m * a.x;
        m = l.y + r.y; m = (m > 0.0f) ? m : NEG_SLOPE * m; s += m * a.y;
        m = l.z + r.z; m = (m > 0.0f) ? m : NEG_SLOPE * m; s += m * a.z;
        m = l.w + r.w; m = (m > 0.0f) ? m : NEG_SLOPE * m; s += m * a.w;
    }
    escore[t] = s;
    atomic_max_f32(&emax[dst * H_HEADS + h], s);
}

__global__ void k_edge_exp(const int* __restrict__ ei,
                           float* __restrict__ escore,
                           const float* __restrict__ emax,
                           float* __restrict__ denom) {
    int t = blockIdx.x * blockDim.x + threadIdx.x;
    if (t >= ETOT * H_HEADS) return;
    int e = t >> 2, h = t & 3;
    int src, dst; edge_endpoints(e, ei, src, dst);
    float ex = expf(escore[t] - emax[dst * H_HEADS + h]);
    escore[t] = ex;
    atomicAdd(&denom[dst * H_HEADS + h], ex);
}

__global__ void k_edge_agg(const int* __restrict__ ei,
                           const float* __restrict__ escore,
                           const float* __restrict__ denom,
                           const float* __restrict__ xl,
                           float* __restrict__ agg) {
    int t = blockIdx.x * blockDim.x + threadIdx.x;
    if (t >= ETOT * H_HEADS) return;
    int e = t >> 2, h = t & 3;
    int src, dst; edge_endpoints(e, ei, src, dst);
    float alpha = escore[t] / denom[dst * H_HEADS + h];
    const float* xls = xl + (size_t)src * HC_DIM + h * C_DIM;
    float* out = agg + (size_t)dst * HC_DIM + h * C_DIM;
#pragma unroll
    for (int c = 0; c < C_DIM; ++c)
        atomicAdd(&out[c], alpha * xls[c]);
}

__global__ void k_gat_bias(float* __restrict__ agg,
                           const float* __restrict__ gat_bias) {
    int t = blockIdx.x * blockDim.x + threadIdx.x;
    if (t < N_NODES * HC_DIM) agg[t] += gat_bias[t & (HC_DIM - 1)];
}

// ---------------- GRU elementwise (h0 = 0) ----------------
__global__ void k_gru(const float* __restrict__ gi,
                      const float* __restrict__ b_hh,
                      float* __restrict__ hout) {
    int t = blockIdx.x * blockDim.x + threadIdx.x;
    if (t >= N_NODES * GH_DIM) return;
    int i = t / GH_DIM, j = t % GH_DIM;
    const float* g = gi + (size_t)i * (3 * GH_DIM);
    float r = 1.0f / (1.0f + expf(-(g[j]              + b_hh[j])));
    float z = 1.0f / (1.0f + expf(-(g[GH_DIM + j]     + b_hh[GH_DIM + j])));
    float n = tanhf(g[2 * GH_DIM + j] + r * b_hh[2 * GH_DIM + j]);
    hout[t] = (1.0f - z) * n;
}

// ---------------- std head ----------------
__global__ void k_std(const float* __restrict__ log_std, float* __restrict__ out_std) {
    int i = blockIdx.x * blockDim.x + threadIdx.x;
    if (i >= N_NODES) return;
    float v = log_std[i];
    v = fminf(fmaxf(v, -10.0f), 2.0f);
    out_std[i] = expf(v);
}

// ---------------- host side ----------------
extern "C" void kernel_launch(void* const* d_in, const int* in_sizes, int n_in,
                              void* d_out, int out_size, void* d_ws, size_t ws_size,
                              hipStream_t stream) {
    (void)in_sizes; (void)n_in; (void)out_size; (void)ws_size;

    const float* x        = (const float*)d_in[0];
    const int*   ei       = (const int*)  d_in[1];   // [2, E]
    const float* W_l      = (const float*)d_in[2];
    const float* b_l      = (const float*)d_in[3];
    const float* W_r      = (const float*)d_in[4];
    const float* b_r      = (const float*)d_in[5];
    const float* att      = (const float*)d_in[6];
    const float* gat_bias = (const float*)d_in[7];
    const float* W_ih     = (const float*)d_in[8];
    const float* b_ih     = (const float*)d_in[10];
    const float* b_hh     = (const float*)d_in[11];
    const float* Wp       = (const float*)d_in[12];
    const float* bp       = (const float*)d_in[13];
    const float* log_std  = (const float*)d_in[14];

    float* out_mean = (float*)d_out;                        // [N, N]
    float* out_std  = out_mean + (size_t)N_NODES * N_NODES; // [N]

    // workspace carve-up: float region, then bf16 region (16B aligned)
    float* ws = (float*)d_ws;
    float* xl     = ws;                                      // N*HC
    float* xr     = xl     + (size_t)N_NODES * HC_DIM;
    float* escore = xr     + (size_t)N_NODES * HC_DIM;       // ETOT*H
    float* emax   = escore + (size_t)ETOT * H_HEADS;         // N*H
    float* denom  = emax   + (size_t)N_NODES * H_HEADS;      // N*H
    float* agg    = denom  + (size_t)N_NODES * H_HEADS;      // N*HC
    float* gi     = agg    + (size_t)N_NODES * HC_DIM;       // N*3*GH
    float* hbuf   = gi     + (size_t)N_NODES * 3 * GH_DIM;   // N*GH
    bf16_t* h_hi  = (bf16_t*)(hbuf + (size_t)N_NODES * GH_DIM);
    bf16_t* h_lo  = h_hi + (size_t)N_NODES * GH_DIM;
    bf16_t* wp_hi = h_lo + (size_t)N_NODES * GH_DIM;
    bf16_t* wp_lo = wp_hi + (size_t)N_NODES * GH_DIM;

    const int THREADS = 256;
    const int mBlk = (N_NODES + 63) / 64;  // 79
    const int nHW  = N_NODES * GH_DIM;     // h / Wp element count (both [N,GH])

    // 0) init accumulators
    k_init<<<(N_NODES * HC_DIM + THREADS - 1) / THREADS, THREADS, 0, stream>>>(emax, denom, agg);

    // 1) xl = x @ W_l + b_l ; xr = x @ W_r + b_r   (exact f32 WMMA, NN)
    k_gemm_wmma_f32<false><<<dim3(mBlk, 2), THREADS, 0, stream>>>(x, W_l, b_l, xl,
                                                                  N_NODES, HC_DIM, F_DIM);
    k_gemm_wmma_f32<false><<<dim3(mBlk, 2), THREADS, 0, stream>>>(x, W_r, b_r, xr,
                                                                  N_NODES, HC_DIM, F_DIM);

    // 2) edge attention: scores + segment softmax + weighted scatter
    const int eBlocks = (ETOT * H_HEADS + THREADS - 1) / THREADS;
    k_edge_score<<<eBlocks, THREADS, 0, stream>>>(ei, xl, xr, att, escore, emax);
    k_edge_exp  <<<eBlocks, THREADS, 0, stream>>>(ei, escore, emax, denom);
    k_edge_agg  <<<eBlocks, THREADS, 0, stream>>>(ei, escore, denom, xl, agg);
    k_gat_bias  <<<(N_NODES * HC_DIM + THREADS - 1) / THREADS, THREADS, 0, stream>>>(agg, gat_bias);

    // 3) GRU input GEMM: gi = gat_out @ W_ih^T + b_ih   (exact f32 WMMA, NT)
    k_gemm_wmma_f32<true><<<dim3(mBlk, (3 * GH_DIM + 63) / 64), THREADS, 0, stream>>>(
        agg, W_ih, b_ih, gi, N_NODES, 3 * GH_DIM, HC_DIM);

    // 4) GRU elementwise -> h
    k_gru<<<(N_NODES * GH_DIM + THREADS - 1) / THREADS, THREADS, 0, stream>>>(gi, b_hh, hbuf);

    // 5) policy head via bf16x3: split h and Wp, then mean = h @ Wp^T + bp
    k_split_bf16<<<(nHW + THREADS - 1) / THREADS, THREADS, 0, stream>>>(hbuf, h_hi, h_lo, nHW);
    k_split_bf16<<<(nHW + THREADS - 1) / THREADS, THREADS, 0, stream>>>(Wp,   wp_hi, wp_lo, nHW);
    k_gemm_bf16x3_nt<<<dim3(mBlk, mBlk), THREADS, 0, stream>>>(
        h_hi, h_lo, wp_hi, wp_lo, bp, out_mean, N_NODES, N_NODES, GH_DIM);

    // 6) std = exp(clip(log_std, -10, 2))
    k_std<<<(N_NODES + THREADS - 1) / THREADS, THREADS, 0, stream>>>(log_std, out_std);
}
